// PointTransformerLayer_77687368450489
// MI455X (gfx1250) — matrix operand order repaired and checked by reading, not compiled, not run on hardware
//
#include <hip/hip_runtime.h>

// Problem constants (fixed by the reference: B=1, N=1024, DIM=64, EXP=4, K=16)
#define NPTS 1024
#define DIMV 64
#define HID  256
#define KNN  16

typedef unsigned short u16;
typedef __bf16 bf16_t;
typedef bf16_t v16bf __attribute__((ext_vector_type(16)));
typedef bf16_t v8bf  __attribute__((ext_vector_type(8)));
typedef float  v8f   __attribute__((ext_vector_type(8)));

// ---------- helpers ----------

// Native f32 -> bf16 convert (single VALU op on gfx1250, RTNE)
__device__ __forceinline__ u16 f2bf(float f) {
  return __builtin_bit_cast(u16, (bf16_t)f);
}

// Low 32 bits of a generic pointer to __shared__ = LDS byte offset (ISA 10.2 aperture rules).
__device__ __forceinline__ unsigned lds_off(const void* p) {
  return (unsigned)(unsigned long long)p;
}

// Per-lane async global->LDS copy of 16 bytes (GLOBAL_LOAD_ASYNC_TO_LDS_B128, ASYNCcnt).
__device__ __forceinline__ void async_copy_b128(unsigned lds_addr, const void* gptr) {
  asm volatile("global_load_async_to_lds_b128 %0, %1, off"
               :: "v"(lds_addr), "v"(gptr)
               : "memory");
}
__device__ __forceinline__ void async_wait0() {
  asm volatile("s_wait_asynccnt 0x0" ::: "memory");
}

// A-fragment: 16x32 bf16 tile, row-major, leading dim lda (elements).
// Per ISA 7.12.2 a lane owns two contiguous 8-element K-runs -> two b128 loads.
__device__ __forceinline__ v16bf load_a_frag(const u16* __restrict__ A, int lda) {
  const int lane = threadIdx.x & 31;
  const int half = lane >> 4;
  const int m    = lane & 15;
  const u16* p = A + m * lda + (half << 3);
  v8bf lo = *(const v8bf*)(p);            // k = half*8 + 0..7
  v8bf hi = *(const v8bf*)(p + 16);       // k = 16 + half*8 + 0..7
  return __builtin_shufflevector(lo, hi, 0, 1, 2, 3, 4, 5, 6, 7,
                                 8, 9, 10, 11, 12, 13, 14, 15);
}

// B-fragment straight from the pre-transposed weight BT[n*K + k]:
// lane (half,n) needs B[k = c*32 + half*8 + t][tn+n] and B[k = c*32 + 16 + half*8 + t][tn+n],
// t=0..7 -> two 16-byte runs of BT -> two b128 loads, no LDS, no transpose at use site.
__device__ __forceinline__ v16bf load_b_frag_t(const u16* __restrict__ BT, int K,
                                               int tn, int c) {
  const int lane = threadIdx.x & 31;
  const int half = lane >> 4;
  const int n    = lane & 15;
  const u16* p = BT + (tn + n) * K + c * 32 + (half << 3);
  v8bf lo = *(const v8bf*)(p);
  v8bf hi = *(const v8bf*)(p + 16);
  return __builtin_shufflevector(lo, hi, 0, 1, 2, 3, 4, 5, 6, 7,
                                 8, 9, 10, 11, 12, 13, 14, 15);
}

// ---------- fp32 -> bf16 conversions ----------

__global__ __launch_bounds__(256) void cvt_bf16_kernel(const float* __restrict__ src,
                                                       u16* __restrict__ dst, int n) {
  int gid = blockIdx.x * 256 + threadIdx.x;
  if (gid < n) dst[gid] = f2bf(src[gid]);
}

// Transpose-convert: dst[n*K + k] = bf16(src[k*N + n])  (weights are tiny; one-shot)
template <int KT>
__global__ __launch_bounds__(256) void cvt_bf16_t_kernel(const float* __restrict__ src,
                                                         u16* __restrict__ dst, int total) {
  int gid = blockIdx.x * 256 + threadIdx.x;
  if (gid >= total) return;
  const int n = gid / KT, k = gid % KT;       // KT is a power of two -> shifts
  const int N = total / KT;
  dst[gid] = f2bf(src[k * N + n]);
}

// ---------- generic WMMA GEMM: C = act(A[MxK] * BT^T + bias) ----------
// 4 waves / block; each wave computes a 16 x (NT*16) strip: A fragment loaded
// once per K-chunk and reused across NT column tiles (NT back-to-back WMMAs).

template <bool RELU, bool OUT_BF16, int KT, int NT>
__global__ __launch_bounds__(128) void gemm_bf16_kernel(
    const u16* __restrict__ A, const u16* __restrict__ BT,
    const float* __restrict__ bias, void* __restrict__ Cout,
    int M, int N) {
  (void)M;
  const int wave = threadIdx.x >> 5;
  const int tm   = (blockIdx.x * 4 + wave) << 4;
  const int tn0  = blockIdx.y * (NT * 16);
  v8f zero = {};
  v8f acc[NT];
#pragma unroll
  for (int t = 0; t < NT; ++t) acc[t] = zero;
#pragma unroll
  for (int c = 0; c < KT / 32; ++c) {
    v16bf a = load_a_frag(A + tm * KT + c * 32, KT);
#pragma unroll
    for (int t = 0; t < NT; ++t) {
      v16bf b = load_b_frag_t(BT, KT, tn0 + t * 16, c);
      acc[t] = __builtin_amdgcn_wmma_f32_16x16x32_bf16(false, a, false, b, (short)0,
                                                       acc[t], false, false);
    }
  }
  const int lane = threadIdx.x & 31;
  const int half = lane >> 4, n = lane & 15;
#pragma unroll
  for (int t = 0; t < NT; ++t) {
    const int nc = tn0 + t * 16 + n;
    const float bv = bias ? bias[nc] : 0.0f;
#pragma unroll
    for (int r = 0; r < 8; ++r) {
      const int m = tm + (half << 3) + r;
      float val = acc[t][r] + bv;
      if (RELU) val = fmaxf(val, 0.0f);
      if (OUT_BF16) ((u16*)Cout)[m * N + nc] = f2bf(val);
      else          ((float*)Cout)[m * N + nc] = val;
    }
  }
}

// ---------- kNN: async-stage pos into LDS, then wave-per-query argmin x16 ----------

__global__ __launch_bounds__(256) void knn_kernel(const float* __restrict__ pos,
                                                  int* __restrict__ idx) {
  __shared__ __attribute__((aligned(16))) float s_pos[NPTS * 3];
  // 12 KB identity-layout copy, ASYNCcnt-tracked (8 waves reuse it -> 8x traffic cut)
  for (int t = threadIdx.x; t < NPTS * 3 / 4; t += 256)
    async_copy_b128(lds_off(&s_pos[t * 4]), pos + t * 4);
  async_wait0();
  __syncthreads();

  const int wave = threadIdx.x >> 5;
  const int lane = threadIdx.x & 31;
  const int i    = blockIdx.x * 8 + wave;

  const float px = s_pos[i * 3 + 0], py = s_pos[i * 3 + 1], pz = s_pos[i * 3 + 2];
  float dist[NPTS / 32];
#pragma unroll
  for (int t = 0; t < NPTS / 32; ++t) {
    const int j  = t * 32 + lane;
    const float dx = px - s_pos[j * 3 + 0];
    const float dy = py - s_pos[j * 3 + 1];
    const float dz = pz - s_pos[j * 3 + 2];
    dist[t] = dx * dx + dy * dy + dz * dz;          // squared: same ordering
  }
  unsigned taken = 0u;
  for (int s = 0; s < KNN; ++s) {
    float best = __builtin_inff();
    int   bt   = 0;
#pragma unroll
    for (int t = 0; t < NPTS / 32; ++t)
      if (!((taken >> t) & 1u) && dist[t] < best) { best = dist[t]; bt = t; }
    float rbest = best;
    int   rj    = bt * 32 + lane;
#pragma unroll
    for (int off = 16; off > 0; off >>= 1) {        // lexicographic (dist, j) min
      float od = __shfl_xor(rbest, off, 32);
      int   oj = __shfl_xor(rj, off, 32);
      if (od < rbest || (od == rbest && oj < rj)) { rbest = od; rj = oj; }
    }
    if (lane == 0) idx[i * KNN + s] = rj;
    if ((rj & 31) == lane) taken |= 1u << (rj >> 5); // exclude winner next round
  }
}

// ---------- rel-pos hidden: relu(rel_pos @ P1 + pb1), gathered pairs ----------

__global__ __launch_bounds__(256) void rpe_hidden_kernel(
    const float* __restrict__ pos, const int* __restrict__ idx,
    const float* __restrict__ P1, const float* __restrict__ pb1,
    u16* __restrict__ H1) {
  const int gid = blockIdx.x * 256 + threadIdx.x;   // p*64 + d
  const int p = gid >> 6, d = gid & 63;
  const int i = p >> 4;
  const int j = idx[p];
  const float rx = pos[i * 3 + 0] - pos[j * 3 + 0];
  const float ry = pos[i * 3 + 1] - pos[j * 3 + 1];
  const float rz = pos[i * 3 + 2] - pos[j * 3 + 2];
  const float hv = rx * P1[0 * DIMV + d] + ry * P1[1 * DIMV + d] +
                   rz * P1[2 * DIMV + d] + pb1[d];
  H1[gid] = f2bf(fmaxf(hv, 0.0f));
}

// ---------- rpe GEMM (H1 @ P2 + pb2, full 64-wide strip) + gather epilogue ----------
//   h  = rpe + q_i - k_j   (bf16, feeds A1 GEMM)
//   vg = rpe + v_j         (f32, aggregation operand)

__global__ __launch_bounds__(128) void rpe_fuse_kernel(
    const u16* __restrict__ H1, const u16* __restrict__ P2T,
    const float* __restrict__ pb2, const float* __restrict__ q,
    const float* __restrict__ kmat, const float* __restrict__ v,
    const int* __restrict__ idx, u16* __restrict__ h, float* __restrict__ vg) {
  const int wave = threadIdx.x >> 5;
  const int tm   = (blockIdx.x * 4 + wave) << 4;
  v8f zero = {};
  v8f acc[4];
#pragma unroll
  for (int t = 0; t < 4; ++t) acc[t] = zero;
#pragma unroll
  for (int c = 0; c < DIMV / 32; ++c) {
    v16bf a = load_a_frag(H1 + tm * DIMV + c * 32, DIMV);
#pragma unroll
    for (int t = 0; t < 4; ++t) {
      v16bf b = load_b_frag_t(P2T, DIMV, t * 16, c);
      acc[t] = __builtin_amdgcn_wmma_f32_16x16x32_bf16(false, a, false, b, (short)0,
                                                       acc[t], false, false);
    }
  }
  const int lane = threadIdx.x & 31;
  const int half = lane >> 4, n = lane & 15;
#pragma unroll
  for (int r = 0; r < 8; ++r) {
    const int p = tm + (half << 3) + r;
    const int i = p >> 4;
    const int j = idx[p];                 // one gather per output row, reused x4
#pragma unroll
    for (int t = 0; t < 4; ++t) {
      const int nd = t * 16 + n;
      const float rpe = acc[t][r] + pb2[nd];
      h[p * DIMV + nd]  = f2bf(rpe + q[i * DIMV + nd] - kmat[j * DIMV + nd]);
      vg[p * DIMV + nd] = rpe + v[j * DIMV + nd];
    }
  }
}

// ---------- softmax over K neighbors (per channel) + weighted aggregate ----------
// Stages per-query sim/vg tiles (4 KB each) into LDS via async copy: sim is
// read twice (max + exp), so LDS staging halves L2 traversals.

__global__ __launch_bounds__(64) void softmax_agg_kernel(
    const float* __restrict__ sim, const float* __restrict__ vg,
    float* __restrict__ out) {
  __shared__ __attribute__((aligned(16))) float s_sim[KNN * DIMV];
  __shared__ __attribute__((aligned(16))) float s_vg[KNN * DIMV];
  const int i = blockIdx.x;
  const float* gs = sim + (size_t)i * KNN * DIMV;
  const float* gv = vg + (size_t)i * KNN * DIMV;
  for (int t = threadIdx.x; t < KNN * DIMV / 4; t += 64) {
    async_copy_b128(lds_off(&s_sim[t * 4]), gs + t * 4);
    async_copy_b128(lds_off(&s_vg[t * 4]), gv + t * 4);
  }
  async_wait0();
  __syncthreads();

  const int d = threadIdx.x;
  float m = -__builtin_inff();
#pragma unroll
  for (int kk = 0; kk < KNN; ++kk)
    m = fmaxf(m, s_sim[kk * DIMV + d]);
  float Z = 0.0f, acc = 0.0f;
#pragma unroll
  for (int kk = 0; kk < KNN; ++kk) {
    const float e = __expf(s_sim[kk * DIMV + d] - m);
    Z += e;
    acc += e * s_vg[kk * DIMV + d];
  }
  out[i * DIMV + d] = acc / Z;
}

// ---------- host orchestration ----------

extern "C" void kernel_launch(void* const* d_in, const int* in_sizes, int n_in,
                              void* d_out, int out_size, void* d_ws, size_t ws_size,
                              hipStream_t stream) {
  (void)in_sizes; (void)n_in; (void)out_size; (void)ws_size;
  const float* x   = (const float*)d_in[0];
  const float* pos = (const float*)d_in[1];
  const float* Wq  = (const float*)d_in[2];
  const float* Wk  = (const float*)d_in[3];
  const float* Wv  = (const float*)d_in[4];
  const float* P1  = (const float*)d_in[5];
  const float* pb1 = (const float*)d_in[6];
  const float* P2  = (const float*)d_in[7];
  const float* pb2 = (const float*)d_in[8];
  const float* A1  = (const float*)d_in[9];
  const float* ab1 = (const float*)d_in[10];
  const float* A2  = (const float*)d_in[11];
  const float* ab2 = (const float*)d_in[12];
  float* out = (float*)d_out;

  char* ws = (char*)d_ws;
  size_t off = 0;
  auto alloc = [&](size_t bytes) {
    char* p = ws + off;
    off += (bytes + 255) & ~(size_t)255;
    return p;
  };
  u16*   xb  = (u16*)alloc((size_t)NPTS * DIMV * 2);
  u16*   WqT = (u16*)alloc((size_t)DIMV * DIMV * 2);
  u16*   WkT = (u16*)alloc((size_t)DIMV * DIMV * 2);
  u16*   WvT = (u16*)alloc((size_t)DIMV * DIMV * 2);
  u16*   P2T = (u16*)alloc((size_t)DIMV * DIMV * 2);
  u16*   A1T = (u16*)alloc((size_t)DIMV * HID * 2);   // [HID, DIMV] transposed
  u16*   A2T = (u16*)alloc((size_t)HID * DIMV * 2);   // [DIMV, HID] transposed
  float* qf  = (float*)alloc((size_t)NPTS * DIMV * 4);
  float* kf  = (float*)alloc((size_t)NPTS * DIMV * 4);
  float* vf  = (float*)alloc((size_t)NPTS * DIMV * 4);
  int*   idx = (int*)alloc((size_t)NPTS * KNN * 4);
  u16*   H1  = (u16*)alloc((size_t)NPTS * KNN * DIMV * 2);
  u16*   hb  = (u16*)alloc((size_t)NPTS * KNN * DIMV * 2);
  float* vg  = (float*)alloc((size_t)NPTS * KNN * DIMV * 4);
  u16*   H2  = (u16*)alloc((size_t)NPTS * KNN * HID * 2);
  float* sim = (float*)alloc((size_t)NPTS * KNN * DIMV * 4);

  // 1. fp32 -> bf16; weights transposed so B fragments are contiguous K runs
  cvt_bf16_kernel<<<(NPTS * DIMV + 255) / 256, 256, 0, stream>>>(x, xb, NPTS * DIMV);
  cvt_bf16_t_kernel<DIMV><<<(DIMV * DIMV + 255) / 256, 256, 0, stream>>>(Wq, WqT, DIMV * DIMV);
  cvt_bf16_t_kernel<DIMV><<<(DIMV * DIMV + 255) / 256, 256, 0, stream>>>(Wk, WkT, DIMV * DIMV);
  cvt_bf16_t_kernel<DIMV><<<(DIMV * DIMV + 255) / 256, 256, 0, stream>>>(Wv, WvT, DIMV * DIMV);
  cvt_bf16_t_kernel<DIMV><<<(DIMV * DIMV + 255) / 256, 256, 0, stream>>>(P2, P2T, DIMV * DIMV);
  cvt_bf16_t_kernel<DIMV><<<(DIMV * HID + 255) / 256, 256, 0, stream>>>(A1, A1T, DIMV * HID);
  cvt_bf16_t_kernel<HID><<<(HID * DIMV + 255) / 256, 256, 0, stream>>>(A2, A2T, HID * DIMV);

  // 2. q, k, v projections (WMMA, one 16x64 strip per wave)
  dim3 g_qkv(NPTS / 64, 1);
  gemm_bf16_kernel<false, false, DIMV, 4><<<g_qkv, 128, 0, stream>>>(xb, WqT, nullptr, qf, NPTS, DIMV);
  gemm_bf16_kernel<false, false, DIMV, 4><<<g_qkv, 128, 0, stream>>>(xb, WkT, nullptr, kf, NPTS, DIMV);
  gemm_bf16_kernel<false, false, DIMV, 4><<<g_qkv, 128, 0, stream>>>(xb, WvT, nullptr, vf, NPTS, DIMV);

  // 3. kNN selection (async LDS staging; matches top_k(-dist) ordering)
  knn_kernel<<<NPTS / 8, 256, 0, stream>>>(pos, idx);

  // 4. rel-pos MLP hidden layer (3 -> 64, VALU) on gathered pairs
  rpe_hidden_kernel<<<(NPTS * KNN * DIMV) / 256, 256, 0, stream>>>(pos, idx, P1, pb1, H1);

  // 5. rel-pos MLP output GEMM (WMMA) fused with gather of q/k/v
  rpe_fuse_kernel<<<NPTS * KNN / 64, 128, 0, stream>>>(H1, P2T, pb2, qf, kf, vf, idx, hb, vg);

  // 6. attention MLP layer 1: relu(h @ A1 + ab1) -> bf16 (WMMA, dominant GEMM)
  dim3 g_g1(NPTS * KNN / 64, HID / 64);
  gemm_bf16_kernel<true, true, DIMV, 4><<<g_g1, 128, 0, stream>>>(hb, A1T, ab1, H2, NPTS * KNN, HID);

  // 7. attention MLP layer 2: sim = H2 @ A2 + ab2 (WMMA, K=256 fully unrolled)
  dim3 g_g2(NPTS * KNN / 64, 1);
  gemm_bf16_kernel<false, false, HID, 4><<<g_g2, 128, 0, stream>>>(H2, A2T, ab2, sim, NPTS * KNN, DIMV);

  // 8. per-channel softmax over neighbors + aggregation (async LDS staging)
  softmax_agg_kernel<<<NPTS, 64, 0, stream>>>(sim, vg, out);
}
